// MultiHeadAttention_9912784519482
// MI455X (gfx1250) — matrix operand order repaired and checked by reading, not compiled, not run on hardware
//
#include <hip/hip_runtime.h>

// ---------------------------------------------------------------------------
// MultiHeadAttention forward for MI455X (gfx1250), wave32 + WMMA bf16.
// B=2, S=2048, D=1024, H=16, HD=64.  Pipeline:
//   1) f32->bf16 casts of x and the four weight matrices
//   2) Q/K/V projections via v_wmma_f32_16x16x32_bf16 (wave tile 32x64);
//      V stored transposed [bh,hd,s]
//   3) flash-attention: K/V tiles DMA'd to LDS with global_load_async_to_lds_b128
//      (double-buffered, s_wait_asynccnt), online softmax, P via LDS re-layout
//   4) output projection + bias, fp32 result
// Workspace (48 MiB): 0 xbf | 8Mi Wq | 10Mi Wk | 12Mi Wv | 14Mi Wo |
//                     16Mi Q | 24Mi K | 32Mi V^T | 40Mi ctx
// ---------------------------------------------------------------------------

#define Bb 2
#define Ss 2048
#define Dd 1024
#define Hh 16
#define HDd 64

typedef __attribute__((ext_vector_type(16))) __bf16 v16bf;
typedef __attribute__((ext_vector_type(8)))  __bf16 v8bf;
typedef __attribute__((ext_vector_type(8)))  float  v8f;

union FragBF { v16bf v; v8bf h[2]; };

static __device__ inline v8f wmma_bf16(const v16bf& a, const v16bf& b, const v8f& c) {
  return __builtin_amdgcn_wmma_f32_16x16x32_bf16(false, a, false, b, (short)0, c, false, false);
}

static __device__ inline unsigned pack_bf2(float a, float b) {
  unsigned short ua = __builtin_bit_cast(unsigned short, (__bf16)a);
  unsigned short ub = __builtin_bit_cast(unsigned short, (__bf16)b);
  return (unsigned)ua | ((unsigned)ub << 16);
}

// ASYNCcnt-tracked global->LDS DMA (gfx1250). VDST = per-lane LDS byte addr,
// VADDR = per-lane 64-bit global addr.
static __device__ inline void async_g2l_b128(unsigned lds_addr, const void* gaddr) {
  asm volatile("global_load_async_to_lds_b128 %0, %1, off"
               :: "v"(lds_addr), "v"((unsigned long long)(uintptr_t)gaddr) : "memory");
}
#define WAIT_ASYNC(imm) asm volatile("s_wait_asynccnt " #imm ::: "memory")

// ------------------------------ 1) f32 -> bf16 ------------------------------
__global__ void __launch_bounds__(256) cast_f32_bf16(const float* __restrict__ in,
                                                     __bf16* __restrict__ out, int n) {
  int i = (blockIdx.x * 256 + threadIdx.x) * 8;
  if (i + 8 <= n) {
    float4 a = *(const float4*)(in + i);
    float4 b = *(const float4*)(in + i + 4);
    v8bf o;
    o[0] = (__bf16)a.x; o[1] = (__bf16)a.y; o[2] = (__bf16)a.z; o[3] = (__bf16)a.w;
    o[4] = (__bf16)b.x; o[5] = (__bf16)b.y; o[6] = (__bf16)b.z; o[7] = (__bf16)b.w;
    *(v8bf*)(out + i) = o;
  }
}

// --------------------- 2) QKV projection GEMM (bf16 WMMA) -------------------
// C[4096,1024] = X * W^T.  Block = 4 waves; wave tile 32x64 (two 16-row tiles
// sharing each B fragment -> 8 WMMA per 12 b128 loads per K-step).
// mode 0: out[(b*H+h)*S + s][hd] (Q,K) ; mode 1: out[(b*H+h)*HD + hd][s] (V^T)
__global__ void __launch_bounds__(128) qkv_gemm(const __bf16* __restrict__ X,
                                                const __bf16* __restrict__ W,
                                                __bf16* __restrict__ out, int mode) {
  const int lane = threadIdx.x & 31, wave = threadIdx.x >> 5;
  const int mBase = blockIdx.y * 128 + wave * 32;
  const int nBase = blockIdx.x * 64;
  const int col = lane & 15;
  const int koff8 = (lane >> 4) * 8, koff16 = (lane >> 4) * 16;

  v8f acc0[4] = {}, acc1[4] = {};
  for (int k0 = 0; k0 < Dd; k0 += 32) {
    FragBF a0, a1;
    const __bf16* ap0 = X + (size_t)(mBase + col) * Dd + k0;
    const __bf16* ap1 = X + (size_t)(mBase + 16 + col) * Dd + k0;
    a0.h[0] = *(const v8bf*)(ap0 + koff8);
    a0.h[1] = *(const v8bf*)(ap0 + 16 + koff8);
    a1.h[0] = *(const v8bf*)(ap1 + koff8);
    a1.h[1] = *(const v8bf*)(ap1 + 16 + koff8);
#pragma unroll
    for (int t = 0; t < 4; ++t) {
      const __bf16* bp = W + (size_t)(nBase + 16 * t + col) * Dd + k0 + koff8;
      FragBF bfr;
      bfr.h[0] = *(const v8bf*)bp;
      bfr.h[1] = *(const v8bf*)(bp + 16);
      acc0[t] = wmma_bf16(a0.v, bfr.v, acc0[t]);
      acc1[t] = wmma_bf16(a1.v, bfr.v, acc1[t]);
    }
  }
  const int rbase = (lane >> 4) * 8;
#pragma unroll
  for (int t = 0; t < 4; ++t) {
    const int n = nBase + 16 * t + col;
    const int h = n >> 6, hd = n & 63;
#pragma unroll
    for (int r = 0; r < 8; ++r) {
#pragma unroll
      for (int half = 0; half < 2; ++half) {
        const int m = mBase + 16 * half + rbase + r;
        const int b = m >> 11, s = m & (Ss - 1);
        size_t idx = (mode == 0)
                   ? (((size_t)(b * Hh + h) * Ss + s) * HDd + hd)
                   : (((size_t)(b * Hh + h) * HDd + hd) * Ss + s);
        out[idx] = (__bf16)(half ? acc1[t][r] : acc0[t][r]);
      }
    }
  }
}

// ------------------- 3) flash attention (online softmax) --------------------
// grid (S/64, H, B); 4 waves/block; wave owns 16 query rows; KV step 32.
// K (32x64) and V^T (64x32) tiles are shared by all waves: DMA'd to LDS via
// global_load_async_to_lds_b128 (4 instr/wave/tile), double-buffered.
__global__ void __launch_bounds__(128) flash_attn(const __bf16* __restrict__ Q,
                                                  const __bf16* __restrict__ K,
                                                  const __bf16* __restrict__ Vt,
                                                  __bf16* __restrict__ Ctx) {
  __shared__ __bf16 Kt[2][32 * 64];   // [kv][hd]
  __shared__ __bf16 Vl[2][64 * 32];   // [hd][kv]
  __shared__ __bf16 Pbuf[4][16 * 32]; // per-wave P re-layout
  const int lane = threadIdx.x & 31, wave = threadIdx.x >> 5;
  const int tid = threadIdx.x;
  const int qBase = blockIdx.x * 64 + wave * 16;
  const int bh = blockIdx.z * Hh + blockIdx.y;
  const __bf16* Qp = Q + (size_t)bh * Ss * HDd;
  const __bf16* Kp = K + (size_t)bh * Ss * HDd;
  const __bf16* Vp = Vt + (size_t)bh * HDd * Ss;
  const int col = lane & 15;
  const int koff8 = (lane >> 4) * 8, koff16 = (lane >> 4) * 16;
  const float SC = 0.125f;               // 1/sqrt(64)
  const float L2E = 1.44269504088896f;

  // issue one 8KB tile (K 4KB + V 4KB) as 4 async-b128 instructions per wave
  auto issue_tile = [&](int kv0, int buf) {
#pragma unroll
    for (int j = 0; j < 2; ++j) {
      const int c = tid + j * 128;                       // 16B chunk id, 0..255
      const unsigned ldsK = (unsigned)(uintptr_t)(&Kt[buf][0]) + c * 16;
      async_g2l_b128(ldsK, Kp + (size_t)(kv0 + (c >> 3)) * HDd + (c & 7) * 8);
      const unsigned ldsV = (unsigned)(uintptr_t)(&Vl[buf][0]) + c * 16;
      async_g2l_b128(ldsV, Vp + (size_t)(c >> 2) * Ss + kv0 + (c & 3) * 8);
    }
  };

  // Q^T as B-operand: lane holds query column qBase+col; two K-steps over hd.
  FragBF qb[2];
#pragma unroll
  for (int ks = 0; ks < 2; ++ks) {
    const __bf16* p = Qp + (size_t)(qBase + col) * HDd + 32 * ks + koff8;
    qb[ks].h[0] = *(const v8bf*)p;
    qb[ks].h[1] = *(const v8bf*)(p + 16);
  }

  v8f acc[4] = {};
  float run_m = -__builtin_inff(), run_l = 0.f;

  issue_tile(0, 0);
  for (int it = 0; it < Ss / 32; ++it) {
    const int buf = it & 1;
    if (it + 1 < Ss / 32) {
      issue_tile((it + 1) * 32, (it + 1) & 1);
      WAIT_ASYNC(0x4);                 // current tile's 4 DMAs retired (in-order)
    } else {
      WAIT_ASYNC(0x0);
    }
    __syncthreads();                   // all waves' DMAs visible

    // s^T tiles from LDS: s0 = K[0..15] x Q^T, s1 = K[16..31] x Q^T
    v8f s0 = {}, s1 = {};
#pragma unroll
    for (int ks = 0; ks < 2; ++ks) {
      FragBF ka, kb;
      const __bf16* p0 = &Kt[buf][(size_t)col * HDd + 32 * ks];
      ka.h[0] = *(const v8bf*)(p0 + koff8);
      ka.h[1] = *(const v8bf*)(p0 + 16 + koff8);
      s0 = wmma_bf16(ka.v, qb[ks].v, s0);
      const __bf16* p1 = &Kt[buf][(size_t)(16 + col) * HDd + 32 * ks];
      kb.h[0] = *(const v8bf*)(p1 + koff8);
      kb.h[1] = *(const v8bf*)(p1 + 16 + koff8);
      s1 = wmma_bf16(kb.v, qb[ks].v, s1);
    }
    s0 *= SC; s1 *= SC;
    // per-query online softmax (lane pair L, L^16 share a query column)
    float mx = s0[0];
#pragma unroll
    for (int r = 1; r < 8; ++r) mx = fmaxf(mx, s0[r]);
#pragma unroll
    for (int r = 0; r < 8; ++r) mx = fmaxf(mx, s1[r]);
    mx = fmaxf(mx, __shfl_xor(mx, 16, 32));
    const float nm = fmaxf(run_m, mx);
    const float alpha = exp2f((run_m - nm) * L2E);
    run_m = nm;
    float p0v[8], p1v[8], ssum = 0.f;
#pragma unroll
    for (int r = 0; r < 8; ++r) {
      p0v[r] = exp2f((s0[r] - nm) * L2E);
      p1v[r] = exp2f((s1[r] - nm) * L2E);
      ssum += p0v[r] + p1v[r];
    }
    ssum += __shfl_xor(ssum, 16, 32);
    run_l = run_l * alpha + ssum;
    // store P^T into LDS as P[q][kv] bf16 (per-wave buffer)
    __bf16* pb = &Pbuf[wave][(size_t)col * 32];
    const int kvb = (lane >> 4) * 8;
#pragma unroll
    for (int r = 0; r < 8; r += 2) {
      *(unsigned*)(pb + kvb + r)      = pack_bf2(p0v[r], p0v[r + 1]);
      *(unsigned*)(pb + 16 + kvb + r) = pack_bf2(p1v[r], p1v[r + 1]);
    }
    asm volatile("s_wait_dscnt 0" ::: "memory");
    // rescale running context by alpha (per-q, fetched cross-lane)
#pragma unroll
    for (int r = 0; r < 8; ++r) {
      const float a = __shfl(alpha, ((lane >> 4) << 3) + r, 32);
#pragma unroll
      for (int t = 0; t < 4; ++t) acc[t][r] *= a;
    }
    // reload P as WMMA A-operand
    FragBF pf;
    const __bf16* pl = &Pbuf[wave][(size_t)col * 32];
    pf.h[0] = *(const v8bf*)(pl + koff8);
    pf.h[1] = *(const v8bf*)(pl + 16 + koff8);
    // ctx += P x V  (V tile in LDS, lane's hd column contiguous in kv)
#pragma unroll
    for (int t = 0; t < 4; ++t) {
      const __bf16* vp = &Vl[buf][(size_t)(16 * t + col) * 32 + koff8];
      FragBF vf;
      vf.h[0] = *(const v8bf*)vp;
      vf.h[1] = *(const v8bf*)(vp + 16);
      acc[t] = wmma_bf16(pf.v, vf.v, acc[t]);
    }
    __syncthreads();                   // reads done before next overwrite
  }
  // normalize and write ctx[b][s][h*64+hd] as bf16
  const int rbase = (lane >> 4) * 8;
#pragma unroll
  for (int r = 0; r < 8; ++r) {
    const float l = __shfl(run_l, rbase + r, 32);
    const float inv = 1.0f / l;
    const int m = qBase + rbase + r;
#pragma unroll
    for (int t = 0; t < 4; ++t) {
      const int d = blockIdx.y * 64 + 16 * t + col;
      Ctx[((size_t)blockIdx.z * Ss + m) * Dd + d] = (__bf16)(acc[t][r] * inv);
    }
  }
}

// -------------------- 4) output projection + bias (fp32) --------------------
__global__ void __launch_bounds__(128) out_gemm(const __bf16* __restrict__ Ctx,
                                                const __bf16* __restrict__ Wo,
                                                const float* __restrict__ bo,
                                                float* __restrict__ out) {
  const int lane = threadIdx.x & 31, wave = threadIdx.x >> 5;
  const int mBase = blockIdx.y * 128 + wave * 32;
  const int nBase = blockIdx.x * 64;
  const int col = lane & 15;
  const int koff8 = (lane >> 4) * 8, koff16 = (lane >> 4) * 16;

  v8f acc0[4] = {}, acc1[4] = {};
  for (int k0 = 0; k0 < Dd; k0 += 32) {
    FragBF a0, a1;
    const __bf16* ap0 = Ctx + (size_t)(mBase + col) * Dd + k0;
    const __bf16* ap1 = Ctx + (size_t)(mBase + 16 + col) * Dd + k0;
    a0.h[0] = *(const v8bf*)(ap0 + koff8);
    a0.h[1] = *(const v8bf*)(ap0 + 16 + koff8);
    a1.h[0] = *(const v8bf*)(ap1 + koff8);
    a1.h[1] = *(const v8bf*)(ap1 + 16 + koff8);
#pragma unroll
    for (int t = 0; t < 4; ++t) {
      const __bf16* bp = Wo + (size_t)(nBase + 16 * t + col) * Dd + k0 + koff8;
      FragBF bfr;
      bfr.h[0] = *(const v8bf*)bp;
      bfr.h[1] = *(const v8bf*)(bp + 16);
      acc0[t] = wmma_bf16(a0.v, bfr.v, acc0[t]);
      acc1[t] = wmma_bf16(a1.v, bfr.v, acc1[t]);
    }
  }
  const int rbase = (lane >> 4) * 8;
#pragma unroll
  for (int t = 0; t < 4; ++t) {
    const int n = nBase + 16 * t + col;
    const float bias = bo[n];
#pragma unroll
    for (int r = 0; r < 8; ++r) {
      out[(size_t)(mBase + rbase + r) * Dd + n] = acc0[t][r] + bias;
      out[(size_t)(mBase + 16 + rbase + r) * Dd + n] = acc1[t][r] + bias;
    }
  }
}

// ---------------------------------------------------------------------------
extern "C" void kernel_launch(void* const* d_in, const int* in_sizes, int n_in,
                              void* d_out, int out_size, void* d_ws, size_t ws_size,
                              hipStream_t stream) {
  (void)in_sizes; (void)n_in; (void)out_size; (void)ws_size;
  const float* x  = (const float*)d_in[0];
  const float* Wq = (const float*)d_in[1];
  const float* Wk = (const float*)d_in[2];
  const float* Wv = (const float*)d_in[3];
  const float* Wo = (const float*)d_in[4];
  const float* bo = (const float*)d_in[5];

  char* w = (char*)d_ws;
  __bf16* xbf = (__bf16*)(w);
  __bf16* wqb = (__bf16*)(w + (8ull  << 20));
  __bf16* wkb = (__bf16*)(w + (10ull << 20));
  __bf16* wvb = (__bf16*)(w + (12ull << 20));
  __bf16* wob = (__bf16*)(w + (14ull << 20));
  __bf16* Qb  = (__bf16*)(w + (16ull << 20));
  __bf16* Kb  = (__bf16*)(w + (24ull << 20));
  __bf16* Vtb = (__bf16*)(w + (32ull << 20));
  __bf16* Cx  = (__bf16*)(w + (40ull << 20));

  const int nx = Bb * Ss * Dd;   // 4194304
  const int nw = Dd * Dd;        // 1048576
  cast_f32_bf16<<<nx / (8 * 256), 256, 0, stream>>>(x,  xbf, nx);
  cast_f32_bf16<<<nw / (8 * 256), 256, 0, stream>>>(Wq, wqb, nw);
  cast_f32_bf16<<<nw / (8 * 256), 256, 0, stream>>>(Wk, wkb, nw);
  cast_f32_bf16<<<nw / (8 * 256), 256, 0, stream>>>(Wv, wvb, nw);
  cast_f32_bf16<<<nw / (8 * 256), 256, 0, stream>>>(Wo, wob, nw);

  dim3 gg(Dd / 64, (Bb * Ss) / 128);     // 16 x 32 blocks
  qkv_gemm<<<gg, 128, 0, stream>>>(xbf, wqb, Qb,  0);
  qkv_gemm<<<gg, 128, 0, stream>>>(xbf, wkb, Kb,  0);
  qkv_gemm<<<gg, 128, 0, stream>>>(xbf, wvb, Vtb, 1);

  dim3 gf(Ss / 64, Hh, Bb);              // 32 x 16 x 2 blocks
  flash_attn<<<gf, 128, 0, stream>>>(Qb, Kb, Vtb, Cx);

  out_gemm<<<gg, 128, 0, stream>>>(Cx, wob, bo, (float*)d_out);
}
